// Qwen3VLMoeTextExpertsTransposed_9775345566132
// MI455X (gfx1250) — matrix operand order repaired and checked by reading, not compiled, not run on hardware
//
#include <hip/hip_runtime.h>
#include <hip/hip_bf16.h>
#include <math.h>

// ---- problem constants (match reference) ----
#define E_  32
#define H_  2048
#define I_  768
#define K_  4
#define T_  2048
#define LD2I (2*I_)          // 1536, leading dim of gate_up_proj[e] ([H, 2I] row-major)

#define CAP 8192             // worst-case assignments per expert (T*K)
#define MAX_TILES (CAP/16)   // 512
#define IPAD (I_ + 8)        // padded LDS row stride (776 halfs = 1552B, 16B-aligned, bank-spread)

typedef __bf16 bf16_t;
typedef __attribute__((ext_vector_type(16))) __bf16 v16bf;
typedef __attribute__((ext_vector_type(8)))  __bf16 v8bf;
typedef __attribute__((ext_vector_type(8)))  float  v8f;

// D = A(16x32 bf16) * B(32x16 bf16) + C(16x16 f32)
__device__ __forceinline__ v8f wmma_bf16(v16bf a, v16bf b, v8f c) {
    return __builtin_amdgcn_wmma_f32_16x16x32_bf16(
        /*neg_a=*/false, a, /*neg_b=*/false, b,
        /*c_mod=*/(short)0, c, /*reuse_a=*/false, /*reuse_b=*/false);
}

// Load 8 consecutive f32 and place as bf16 into a[e0..e0+7]
__device__ __forceinline__ void loadA8(const float* __restrict__ p, v16bf& a, int e0) {
    float4 f0 = *(const float4*)(p);
    float4 f1 = *(const float4*)(p + 4);
    a[e0+0] = (bf16_t)f0.x; a[e0+1] = (bf16_t)f0.y;
    a[e0+2] = (bf16_t)f0.z; a[e0+3] = (bf16_t)f0.w;
    a[e0+4] = (bf16_t)f1.x; a[e0+5] = (bf16_t)f1.y;
    a[e0+6] = (bf16_t)f1.z; a[e0+7] = (bf16_t)f1.w;
}

// A fragment (16x32 bf16) for this lane from a global f32 row.
// lanes 0-15: row=lane,    K = {kk+0..7,  kk+16..23}
// lanes 16-31: row=lane-16, K = {kk+8..15, kk+24..31}
__device__ __forceinline__ v16bf loadA_global(const float* __restrict__ rowBase, int kk, int kselA) {
    v16bf a;
    loadA8(rowBase + kk + kselA,      a, 0);
    loadA8(rowBase + kk + kselA + 16, a, 8);
    return a;
}

// B fragment (32x16 bf16): per lane, one weight column, K rows {+0..7, +16..23}.
// p must already point at (k0 + (lane<16?0:8))*ld + column.
__device__ __forceinline__ v16bf loadB_global(const float* __restrict__ p, int ld) {
    v16bf b;
#pragma unroll
    for (int j = 0; j < 16; ++j) b[j] = (bf16_t)p[(size_t)(j < 8 ? j : j + 8) * ld];
    return b;
}

// fast SiLU: x * sigmoid(x), using v_rcp_f32 instead of IEEE division
__device__ __forceinline__ float silu_fast(float x) {
    return x * __builtin_amdgcn_rcpf(1.0f + __expf(-x));
}

// ---------------- kernel 0: zero output + expert counters ----------------
__global__ void moe_zero_kernel(float* __restrict__ out, int n, int* __restrict__ counts) {
    int i = blockIdx.x * blockDim.x + threadIdx.x;
    if (i < n)  out[i] = 0.0f;
    if (i < E_) counts[i] = 0;
}

// ---------------- kernel 1: build per-expert token lists ----------------
__global__ void moe_route_kernel(const int* __restrict__ topk_idx,
                                 const float* __restrict__ topk_w,
                                 int* __restrict__ counts,
                                 int* __restrict__ toks,
                                 float* __restrict__ wts) {
    int t = blockIdx.x * blockDim.x + threadIdx.x;
    if (t >= T_) return;
#pragma unroll
    for (int k = 0; k < K_; ++k) {
        int   e  = topk_idx[t * K_ + k];
        float ww = topk_w  [t * K_ + k];
        int pos = atomicAdd(&counts[e], 1);
        toks[e * CAP + pos] = t;
        wts [e * CAP + pos] = ww;
    }
}

// ---------------- kernel 2: expert GEMMs via WMMA ----------------
// block = 128 threads = 4 wave32 waves; one (expert, 16-token tile) per block.
__global__ __launch_bounds__(128)
void moe_expert_kernel(const float* __restrict__ X,      // [T, H]
                       const float* __restrict__ Wgu,    // [E, H, 2I]
                       const float* __restrict__ Wd,     // [E, I, H]
                       const int*   __restrict__ counts,
                       const int*   __restrict__ toks,
                       const float* __restrict__ wts,
                       float*       __restrict__ out) {  // [T, H]
    const int e    = blockIdx.y;
    const int tile = blockIdx.x;
    const int cnt  = counts[e];
    const int base = tile * 16;
    if (base >= cnt) return;                 // uniform early-exit (before any barrier)

    __shared__ int    sTok[16];
    __shared__ float  sW[16];
    __shared__ bf16_t sInter[16][IPAD];      // [16, I] bf16, padded row stride

    const int tid = threadIdx.x;
    if (tid < 16) {
        int r = base + tid;
        if (r < cnt) { sTok[tid] = toks[e * CAP + r]; sW[tid] = wts[e * CAP + r]; }
        else         { sTok[tid] = 0;                 sW[tid] = 0.0f; }
    }
    __syncthreads();

    const int lane  = tid & 31;
    const int wave  = tid >> 5;              // 0..3
    const int col   = lane & 15;             // fragment row (A) / column (B,C)
    const int kselA = (lane < 16) ? 0 : 8;   // A K-offset selector
    const int kselB = (lane < 16) ? 0 : 8;   // B K-offset selector
    const int rbase = (lane < 16) ? 0 : 8;   // C row base

    const float* aRow = X + (size_t)sTok[col] * H_;           // this lane's token row
    const float* WguE = Wgu + (size_t)e * H_ * LD2I;
    const float* WdE  = Wd  + (size_t)e * I_ * H_;

    // ---- GEMM1: gu[16, 2I] = Xtile[16, H] @ Wgu[H, 2I]; fused SiLU(g)*u -> sInter ----
    // 48 gate/up column-tile pairs; each wave owns 12, processed 2 pairs at a time so
    // one A fragment feeds 4 back-to-back WMMAs.
    for (int i0 = 0; i0 < 12; i0 += 2) {
        const int jpA = wave + 4 * (i0 + 0);
        const int jpB = wave + 4 * (i0 + 1);
        v8f cgA = {}, cuA = {}, cgB = {}, cuB = {};
        const float* bgA = WguE + (size_t)kselB * LD2I + (jpA * 16 + col);
        const float* buA = WguE + (size_t)kselB * LD2I + (I_ + jpA * 16 + col);
        const float* bgB = WguE + (size_t)kselB * LD2I + (jpB * 16 + col);
        const float* buB = WguE + (size_t)kselB * LD2I + (I_ + jpB * 16 + col);
        for (int kk = 0; kk < H_; kk += 32) {
            v16bf a = loadA_global(aRow, kk, kselA);
            const size_t ko = (size_t)kk * LD2I;
            v16bf b0 = loadB_global(bgA + ko, LD2I);
            v16bf b1 = loadB_global(buA + ko, LD2I);
            v16bf b2 = loadB_global(bgB + ko, LD2I);
            v16bf b3 = loadB_global(buB + ko, LD2I);
            cgA = wmma_bf16(a, b0, cgA);
            cuA = wmma_bf16(a, b1, cuA);
            cgB = wmma_bf16(a, b2, cgB);
            cuB = wmma_bf16(a, b3, cuB);
        }
#pragma unroll
        for (int r = 0; r < 8; ++r) {
            sInter[rbase + r][jpA * 16 + col] = (bf16_t)(silu_fast(cgA[r]) * cuA[r]);
            sInter[rbase + r][jpB * 16 + col] = (bf16_t)(silu_fast(cgB[r]) * cuB[r]);
        }
    }
    __syncthreads();

    // ---- GEMM2: out_tile[16, H] = inter[16, I] @ Wd[I, H]; scale by routing w; atomic add ----
    // 128 output column tiles; each wave owns 32, processed 2 at a time (A reused from LDS).
    const bf16_t* arow = &sInter[col][0];
    for (int i0 = 0; i0 < 32; i0 += 2) {
        const int htA = wave + 4 * (i0 + 0);
        const int htB = wave + 4 * (i0 + 1);
        v8f cA = {}, cB = {};
        const float* bbA = WdE + (size_t)kselB * H_ + (htA * 16 + col);
        const float* bbB = WdE + (size_t)kselB * H_ + (htB * 16 + col);
        for (int kk = 0; kk < I_; kk += 32) {
            v8bf lo = *(const v8bf*)(arow + kk + kselA);      // ds_load_b128
            v8bf hi = *(const v8bf*)(arow + kk + kselA + 16); // ds_load_b128
            v16bf a;
#pragma unroll
            for (int j = 0; j < 8; ++j) { a[j] = lo[j]; a[8 + j] = hi[j]; }
            const size_t ko = (size_t)kk * H_;
            v16bf b0 = loadB_global(bbA + ko, H_);
            v16bf b1 = loadB_global(bbB + ko, H_);
            cA = wmma_bf16(a, b0, cA);
            cB = wmma_bf16(a, b1, cB);
        }
#pragma unroll
        for (int r = 0; r < 8; ++r) {
            const int   row = rbase + r;
            const float wr  = sW[row];
            float* orow = out + (size_t)sTok[row] * H_;
            unsafeAtomicAdd(orow + htA * 16 + col, cA[r] * wr);
            unsafeAtomicAdd(orow + htB * 16 + col, cB[r] * wr);
        }
    }
}

extern "C" void kernel_launch(void* const* d_in, const int* in_sizes, int n_in,
                              void* d_out, int out_size, void* d_ws, size_t ws_size,
                              hipStream_t stream) {
    const float* X    = (const float*)d_in[0];   // hidden_states [T, H]
    const int*   idx  = (const int*)  d_in[1];   // top_k_index   [T, K]
    const float* w    = (const float*)d_in[2];   // top_k_weights [T, K]
    const float* Wgu  = (const float*)d_in[3];   // gate_up_proj  [E, H, 2I]
    const float* Wd   = (const float*)d_in[4];   // down_proj     [E, I, H]
    float*       out  = (float*)d_out;           // [T, H]

    // workspace layout: counts | token lists | weight lists  (~2.1 MB)
    int*   counts = (int*)d_ws;
    int*   toks   = (int*)((char*)d_ws + 1024);
    float* wts    = (float*)((char*)d_ws + 1024 + (size_t)E_ * CAP * sizeof(int));

    moe_zero_kernel<<<(T_ * H_ + 255) / 256, 256, 0, stream>>>(out, T_ * H_, counts);
    moe_route_kernel<<<(T_ + 127) / 128, 128, 0, stream>>>(idx, w, counts, toks, wts);
    moe_expert_kernel<<<dim3(MAX_TILES, E_), 128, 0, stream>>>(X, Wgu, Wd, counts, toks, wts, out);
}
